// PointMixerIntraSetLayerPaperv3_20899310862382
// MI455X (gfx1250) — matrix operand order repaired and checked by reading, not compiled, not run on hardware
//
#include <hip/hip_runtime.h>

// ---------------------------------------------------------------------------
// PointMixer intra-set layer, fully fused for MI455X (gfx1250, wave32, WMMA).
// One point per wave: the 16 kNN neighbors are the M=16 rows of every WMMA
// tile.  All GEMM-shaped stages run on v_wmma_f32_16x16x32_f16; only the tiny
// 64->8->8 tail, BN/ReLU epilogues and the neighbor softmax are VALU.
// ---------------------------------------------------------------------------

typedef _Float16 v16h __attribute__((ext_vector_type(16)));
typedef _Float16 v8h  __attribute__((ext_vector_type(8)));
typedef float    v8f  __attribute__((ext_vector_type(8)));

__device__ __forceinline__ v8f zero8() {
  v8f z;
#pragma unroll
  for (int i = 0; i < 8; ++i) z[i] = 0.f;
  return z;
}

__device__ __forceinline__ v8f wmma_f16(v16h a, v16h b, v8f c) {
  // (neg_a, A, neg_b, B, c_mod, C, reuse_a, reuse_b)
  return __builtin_amdgcn_wmma_f32_16x16x32_f16(false, a, false, b, (short)0, c,
                                                false, false);
}

// A fragment of a row-major f16 tile (M=16 rows, K=32 slice starting at kof).
// Contiguous K order (matches B): lane[3:0]=M; lanes 0-15 hold K=0..15(+kof),
// lanes 16-31 hold K=16..31(+kof); VGPR v packs (K,K+1).
__device__ __forceinline__ v16h load_a_frag(const _Float16* tile, int stride,
                                            int lane, int kof) {
  const int row = lane & 15;
  const int hb  = (lane >> 4) * 16;
  const _Float16* r = tile + row * stride + kof;
  v16h a;
#pragma unroll
  for (int e = 0; e < 16; ++e) {
    const int k = hb + e;
    a[e] = r[k];
  }
  return a;
}

// B fragment (K=32 x N=16) from an [N][K]-transposed weight tile in LDS.
// lane[3:0]=N column, lanes 0-15: K=0..15, lanes 16-31: K=16..31 (consecutive
// in the transposed layout -> two aligned 16B ds_load_b128 per fragment).
__device__ __forceinline__ v16h load_b_frag(const _Float16* tileT, int strideK,
                                            int lane, int ntile, int kof) {
  const int ncol = (lane & 15) + ntile * 16;
  const int kb   = (lane >> 4) * 16;
  const _Float16* q = tileT + ncol * strideK + kof + kb;
  v8h lo = *(const v8h*)(q);
  v8h hi = *(const v8h*)(q + 8);
  return __builtin_shufflevector(lo, hi, 0, 1, 2, 3, 4, 5, 6, 7, 8, 9, 10, 11,
                                 12, 13, 14, 15);
}

// B fragment with only K=0..15 valid (upper half zero) — bilinear K=16 step.
__device__ __forceinline__ v16h load_b_frag_k16(const _Float16* tileT, int lane,
                                                int t) {
  v16h b;
  if (lane < 16) {
    const _Float16* q = tileT + (t * 16 + lane) * 16;
    v8h lo = *(const v8h*)(q);
    v8h hi = *(const v8h*)(q + 8);
    b = __builtin_shufflevector(lo, hi, 0, 1, 2, 3, 4, 5, 6, 7, 8, 9, 10, 11,
                                12, 13, 14, 15);
  } else {
#pragma unroll
    for (int e = 0; e < 16; ++e) b[e] = (_Float16)0.f;
  }
  return b;
}

namespace {

struct BlockWeights {                 // staged once per block, [N][K] layouts
  _Float16 W01t[16][96];              // Linear(67->16), K padded to 96
  _Float16 W3t[64][64];               // channelMixMLPs03 Linear(64->64)
  _Float16 WbilT[256][16];            // Wbil[o,i,j] -> [(i*16+o)][j]
  _Float16 W2at[64][32];              // Linear(32->64)
  float Wp2s[3][64];
  float bp2s[64], bn2as[64], bn2ab[64], b3s[64];
  float W2bs[64][8];
  float W2cs[8][8];
  float b01s[16], bbils[16];
  float Wp1s[9], bp1s[3], bnps[3], bnpb[3];
  float bn2bs[8], bn2bb[8], b2cs[8];
};

struct WaveScratch {
  _Float16 A[16][96];                 // [p_r | x[knn] | 0-pad], f16 A-tile
  _Float16 E16[16][32];               // e (then ef) as WMMA A source
  float Ef[16][16];                   // e in f32 for the bilinear contraction
  float Pe[16][64];                   // p_embed (linear_p output)
  float H1[16][64];                   // MLP02 hidden
  float H2[16][8];
  float Wsm[16][8];                   // softmax weights
  float Pr[16][3];
  int   J[16];
};

}  // namespace

#define PTS_PER_BLOCK 2

__global__ __launch_bounds__(64) void pointmixer_fused(
    const float* __restrict__ p, const float* __restrict__ x,
    const int* __restrict__ knn, const float* __restrict__ W01,
    const float* __restrict__ b01, const float* __restrict__ Wbil,
    const float* __restrict__ bbil, const float* __restrict__ Wp1,
    const float* __restrict__ bp1, const float* __restrict__ bnps,
    const float* __restrict__ bnpb, const float* __restrict__ Wp2,
    const float* __restrict__ bp2, const float* __restrict__ W2a,
    const float* __restrict__ s2a, const float* __restrict__ t2a,
    const float* __restrict__ W2b, const float* __restrict__ s2b,
    const float* __restrict__ t2b, const float* __restrict__ W2c,
    const float* __restrict__ b2c, const float* __restrict__ W3,
    const float* __restrict__ b3, float* __restrict__ out, int n) {
  __shared__ alignas(16) BlockWeights wgt;
  __shared__ alignas(16) WaveScratch scr[PTS_PER_BLOCK];

  const int tid = threadIdx.x;

  // ---- stage weights (transposed to [N][K] for vector B-fragment loads) ----
  for (int i = tid; i < 16 * 96; i += 64) {
    const int c = i / 96, k = i % 96;
    wgt.W01t[c][k] = (_Float16)((k < 67) ? W01[k * 16 + c] : 0.f);
  }
  for (int i = tid; i < 64 * 64; i += 64)
    wgt.W3t[i >> 6][i & 63] = (_Float16)W3[(i & 63) * 64 + (i >> 6)];
  for (int i = tid; i < 256 * 16; i += 64) {
    const int cc = i >> 4, j = i & 15;  // cc = i*16 + o
    wgt.WbilT[cc][j] = (_Float16)Wbil[(cc & 15) * 256 + (cc >> 4) * 16 + j];
  }
  for (int i = tid; i < 64 * 32; i += 64)
    wgt.W2at[i >> 5][i & 31] = (_Float16)W2a[(i & 31) * 64 + (i >> 5)];
  for (int i = tid; i < 192; i += 64) (&wgt.Wp2s[0][0])[i] = Wp2[i];
  for (int i = tid; i < 512; i += 64) (&wgt.W2bs[0][0])[i] = W2b[i];
  if (tid < 64) {
    wgt.bp2s[tid] = bp2[tid];
    wgt.bn2as[tid] = s2a[tid];
    wgt.bn2ab[tid] = t2a[tid];
    wgt.b3s[tid] = b3[tid];
    (&wgt.W2cs[0][0])[tid] = W2c[tid];
  }
  if (tid < 16) { wgt.b01s[tid] = b01[tid]; wgt.bbils[tid] = bbil[tid]; }
  if (tid < 9) wgt.Wp1s[tid] = Wp1[tid];
  if (tid < 8) {
    wgt.bn2bs[tid] = s2b[tid]; wgt.bn2bb[tid] = t2b[tid]; wgt.b2cs[tid] = b2c[tid];
  }
  if (tid < 3) { wgt.bp1s[tid] = bp1[tid]; wgt.bnps[tid] = bnps[tid]; wgt.bnpb[tid] = bnpb[tid]; }
  __syncthreads();

  const int wv = tid >> 5;
  const int lane = tid & 31;
  const int pid = blockIdx.x * PTS_PER_BLOCK + wv;
  const bool act = pid < n;
  WaveScratch& S = scr[wv];

  const int col = lane & 15;     // N column of C/D fragments, also kNN row id
  const int hi = lane >> 4;
  const int hi8 = hi * 8;

  const size_t O_XKNN = (size_t)n * 64;
  const size_t O_KNN = O_XKNN + (size_t)n * 16 * 64;
  const size_t O_PR = O_KNN + (size_t)n * 16;

  // ---- gather: p_r, knn copy, A-tile cols 0..2, zero pad cols 67..95 ------
  if (act && lane < 16) {
    const float pcx = p[pid * 3 + 0], pcy = p[pid * 3 + 1], pcz = p[pid * 3 + 2];
    const int j = knn[pid * 16 + lane];
    S.J[lane] = j;
    const float prx = p[j * 3 + 0] - pcx;
    const float pry = p[j * 3 + 1] - pcy;
    const float prz = p[j * 3 + 2] - pcz;
    S.Pr[lane][0] = prx; S.Pr[lane][1] = pry; S.Pr[lane][2] = prz;
    S.A[lane][0] = (_Float16)prx; S.A[lane][1] = (_Float16)pry; S.A[lane][2] = (_Float16)prz;
    const size_t rr = (size_t)pid * 16 + lane;
    out[O_PR + rr * 3 + 0] = prx;
    out[O_PR + rr * 3 + 1] = pry;
    out[O_PR + rr * 3 + 2] = prz;
    out[O_KNN + rr] = (float)j;
  }
  for (int i = lane; i < 16 * 29; i += 32) S.A[i / 29][67 + i % 29] = (_Float16)0.f;
  __syncthreads();

  // ---- gather x[knn] into A-tile cols 3..66 (2 lanes per neighbor row) ----
  if (act) {
    const int j = S.J[col];
    const float4* xb = (const float4*)(x + (size_t)j * 64 + hi * 32);
#pragma unroll
    for (int q = 0; q < 8; ++q) {
      const float4 v = xb[q];
      const int c = 3 + hi * 32 + q * 4;
      S.A[col][c + 0] = (_Float16)v.x; S.A[col][c + 1] = (_Float16)v.y;
      S.A[col][c + 2] = (_Float16)v.z; S.A[col][c + 3] = (_Float16)v.w;
    }
  }
  // ---- linear_p: 3 -> 3(BN,ReLU) -> 64, pure VALU (tiny K) ----------------
  {
    const float prx = S.Pr[col][0], pry = S.Pr[col][1], prz = S.Pr[col][2];
    float q[3];
#pragma unroll
    for (int d = 0; d < 3; ++d) {
      float a = prx * wgt.Wp1s[0 * 3 + d] + pry * wgt.Wp1s[1 * 3 + d] +
                prz * wgt.Wp1s[2 * 3 + d] + wgt.bp1s[d];
      q[d] = fmaxf(a * wgt.bnps[d] + wgt.bnpb[d], 0.f);
    }
#pragma unroll
    for (int cc = 0; cc < 32; ++cc) {
      const int c = hi * 32 + cc;
      S.Pe[col][c] = q[0] * wgt.Wp2s[0][c] + q[1] * wgt.Wp2s[1][c] +
                     q[2] * wgt.Wp2s[2][c] + wgt.bp2s[c];
    }
  }
  __syncthreads();

  // ---- e = ReLU(A(16x67) @ W01 + b01): 3 WMMA K-chunks --------------------
  {
    v8f acc = zero8();
    acc = wmma_f16(load_a_frag(&S.A[0][0], 96, lane, 0),
                   load_b_frag(&wgt.W01t[0][0], 96, lane, 0, 0), acc);
    acc = wmma_f16(load_a_frag(&S.A[0][0], 96, lane, 32),
                   load_b_frag(&wgt.W01t[0][0], 96, lane, 0, 32), acc);
    acc = wmma_f16(load_a_frag(&S.A[0][0], 96, lane, 64),
                   load_b_frag(&wgt.W01t[0][0], 96, lane, 0, 64), acc);
    const float bb = wgt.b01s[col];
#pragma unroll
    for (int v = 0; v < 8; ++v) {
      const float e = fmaxf(acc[v] + bb, 0.f);
      S.Ef[v + hi8][col] = e;
      S.E16[v + hi8][col] = (_Float16)e;
      S.E16[v + hi8][16 + col] = (_Float16)0.f;  // K pad for the bilinear A
    }
  }
  __syncthreads();

  // ---- bilinear e·Wbil·e: 16 WMMAs (K=16) + per-lane i-contraction --------
  v8f e2 = zero8();
  {
    const v16h eA = load_a_frag(&S.E16[0][0], 32, lane, 0);
#pragma unroll
    for (int t = 0; t < 16; ++t) {
      const v16h bt = load_b_frag_k16(&wgt.WbilT[0][0], lane, t);
      const v8f g = wmma_f16(eA, bt, zero8());
#pragma unroll
      for (int v = 0; v < 8; ++v) e2[v] += S.Ef[v + hi8][t] * g[v];
    }
    const float bb = wgt.bbils[col];
#pragma unroll
    for (int v = 0; v < 8; ++v) e2[v] += bb;
  }

  // ---- shrink_p + build ef = [e2 | shrink] (16x32 f16) --------------------
#pragma unroll
  for (int v = 0; v < 8; ++v) {
    const int r = v + hi8;
    const float sh = S.Pe[r][col] + S.Pe[r][16 + col] + S.Pe[r][32 + col] +
                     S.Pe[r][48 + col];
    S.E16[r][col] = (_Float16)e2[v];
    S.E16[r][16 + col] = (_Float16)sh;
  }
  __syncthreads();

  // ---- h1 = ReLU(BN(ef @ W2a)): 4 WMMAs (N tiles) -------------------------
  {
    const v16h efA = load_a_frag(&S.E16[0][0], 32, lane, 0);
#pragma unroll
    for (int nt = 0; nt < 4; ++nt) {
      const v8f h = wmma_f16(efA, load_b_frag(&wgt.W2at[0][0], 32, lane, nt, 0),
                             zero8());
      const int cg = nt * 16 + col;
      const float sa = wgt.bn2as[cg], sb = wgt.bn2ab[cg];
#pragma unroll
      for (int v = 0; v < 8; ++v)
        S.H1[v + hi8][cg] = fmaxf(h[v] * sa + sb, 0.f);
    }
  }
  __syncthreads();

  // ---- tail MLP 64->8->8 (VALU; too small for WMMA) -----------------------
  {
    float h2[4];
#pragma unroll
    for (int ci = 0; ci < 4; ++ci) {
      const int c = hi * 4 + ci;
      float a = 0.f;
#pragma unroll
      for (int kk = 0; kk < 64; ++kk) a += S.H1[col][kk] * wgt.W2bs[kk][c];
      h2[ci] = fmaxf(a * wgt.bn2bs[c] + wgt.bn2bb[c], 0.f);
    }
#pragma unroll
    for (int ci = 0; ci < 4; ++ci) S.H2[col][hi * 4 + ci] = h2[ci];
  }
  __syncthreads();
  {
    float h3[4];
#pragma unroll
    for (int ci = 0; ci < 4; ++ci) {
      const int c = hi * 4 + ci;
      float a = wgt.b2cs[c];
#pragma unroll
      for (int kk = 0; kk < 8; ++kk) a += S.H2[col][kk] * wgt.W2cs[kk][c];
      h3[ci] = a;
    }
    __syncthreads();
#pragma unroll
    for (int ci = 0; ci < 4; ++ci) S.H2[col][hi * 4 + ci] = h3[ci];
  }
  __syncthreads();

  // ---- softmax over the 16 neighbors, per shared-weight channel -----------
  if (lane < 8) {
    float m = -1e30f;
#pragma unroll
    for (int r = 0; r < 16; ++r) m = fmaxf(m, S.H2[r][lane]);
    float s = 0.f;
#pragma unroll
    for (int r = 0; r < 16; ++r) {
      const float e = __expf(S.H2[r][lane] - m);
      S.Wsm[r][lane] = e;
      s += e;
    }
    const float inv = 1.f / s;
#pragma unroll
    for (int r = 0; r < 16; ++r) S.Wsm[r][lane] *= inv;
  }
  __syncthreads();

  // ---- x_v = A[:,3:67] @ W3 (8 WMMAs), fuse +b3 +p_embed, *softmax, store -
  {
    const v16h a0 = load_a_frag(&S.A[0][0], 96, lane, 3);
    const v16h a1 = load_a_frag(&S.A[0][0], 96, lane, 35);
#pragma unroll
    for (int nt = 0; nt < 4; ++nt) {
      v8f acc = wmma_f16(a0, load_b_frag(&wgt.W3t[0][0], 64, lane, nt, 0),
                         zero8());
      acc = wmma_f16(a1, load_b_frag(&wgt.W3t[0][0], 64, lane, nt, 32), acc);
      const int cg = nt * 16 + col;
      const float bv = wgt.b3s[cg];
      float csum = 0.f;
#pragma unroll
      for (int v = 0; v < 8; ++v) {
        const int r = v + hi8;
        const float val = (acc[v] + bv + S.Pe[r][cg]) * S.Wsm[r][cg & 7];
        if (act) out[O_XKNN + ((size_t)pid * 16 + r) * 64 + cg] = val;
        csum += val;
      }
      csum += __shfl_xor(csum, 16, 32);  // combine rows 0-7 with rows 8-15
      if (act && hi == 0) out[(size_t)pid * 64 + cg] = csum;
    }
  }
}

extern "C" void kernel_launch(void* const* d_in, const int* in_sizes, int n_in,
                              void* d_out, int out_size, void* d_ws,
                              size_t ws_size, hipStream_t stream) {
  (void)n_in; (void)out_size; (void)d_ws; (void)ws_size;
  const float* p    = (const float*)d_in[0];
  const float* x    = (const float*)d_in[1];
  const int*   knn  = (const int*)d_in[2];
  const float* W01  = (const float*)d_in[3];
  const float* b01  = (const float*)d_in[4];
  const float* Wbil = (const float*)d_in[5];
  const float* bbil = (const float*)d_in[6];
  const float* Wp1  = (const float*)d_in[7];
  const float* bp1  = (const float*)d_in[8];
  const float* bnps = (const float*)d_in[9];
  const float* bnpb = (const float*)d_in[10];
  const float* Wp2  = (const float*)d_in[11];
  const float* bp2  = (const float*)d_in[12];
  const float* W2a  = (const float*)d_in[13];
  const float* s2a  = (const float*)d_in[14];
  const float* t2a  = (const float*)d_in[15];
  const float* W2b  = (const float*)d_in[16];
  const float* s2b  = (const float*)d_in[17];
  const float* t2b  = (const float*)d_in[18];
  const float* W2c  = (const float*)d_in[19];
  const float* b2c  = (const float*)d_in[20];
  const float* W3   = (const float*)d_in[21];
  const float* b3   = (const float*)d_in[22];
  float* out = (float*)d_out;

  const int n = in_sizes[0] / 3;  // N points
  const int blocks = (n + PTS_PER_BLOCK - 1) / PTS_PER_BLOCK;
  pointmixer_fused<<<blocks, 64, 0, stream>>>(
      p, x, knn, W01, b01, Wbil, bbil, Wp1, bp1, bnps, bnpb, Wp2, bp2, W2a, s2a,
      t2a, W2b, s2b, t2b, W2c, b2c, W3, b3, out, n);
}
